// NNModelEx_63513976373928
// MI455X (gfx1250) — matrix operand order repaired and checked by reading, not compiled, not run on hardware
//
#include <hip/hip_runtime.h>

typedef __attribute__((ext_vector_type(16))) _Float16 v16h;
typedef __attribute__((ext_vector_type(8)))  _Float16 h8;
typedef __attribute__((ext_vector_type(2)))  _Float16 h2;
typedef __attribute__((ext_vector_type(8)))  float    v8f;

union AF { v16h v; h8 c[2]; };

#define TILE_M 128
#define XS_ST  72    // halves; 144B row stride -> 16 distinct banks for 16 rows
#define HS_ST  264   // halves; 528B row stride -> 16 distinct banks for 16 rows

struct Head {
  const float* lw;  const float* lb;
  const float* bew; const float* beb;
  const float* bw;  const float* bb;
  int ridx;
};

// Pure-VALU cross-lane swap via DPP (no ds_bpermute, no DScnt waits).
// CTRL: 0xB1 quad_perm(1,0,3,2)=xor1, 0x4E quad_perm(2,3,0,1)=xor2,
//       0x141 row_half_mirror=xor7, 0x140 row_mirror=xor15.
template <int CTRL>
__device__ __forceinline__ float dpp_swap(float x) {
  return __builtin_bit_cast(
      float, __builtin_amdgcn_update_dpp(0, __builtin_bit_cast(int, x),
                                         CTRL, 0xF, 0xF, true));
}
// 16-lane butterfly sum using XOR masks {1,2,7,15} (spans all 4 lane bits,
// stays inside each DPP row of 16, i.e. inside each half of the wave).
__device__ __forceinline__ float reduce16(float x) {
  x += dpp_swap<0xB1>(x);
  x += dpp_swap<0x4E>(x);
  x += dpp_swap<0x141>(x);
  x += dpp_swap<0x140>(x);
  return x;
}

// A-fragment (16x32 f16): lane&15 = M row; lane>>4 selects K half (+16).
// VGPR pairs cover K = kb+{0..7} then kb+{8..15} -> two 16B LDS loads.
__device__ __forceinline__ void load_a(AF& a, const _Float16* base) {
  a.c[0] = *(const h8*)base;         // K = kb+0..7
  a.c[1] = *(const h8*)(base + 8);   // K = kb+8..15
}
// B-fragment (32x16 f16): lane&15 = N col; lane>>4 selects K half (16).
// VGPR v holds B[kh*16+2v..+1][n] = W[n][kh*16+2v..] -> 16 contiguous halves.
__device__ __forceinline__ void load_b(AF& b, const _Float16* base) {
  b.c[0] = *(const h8*)base;        // K = kh*16+0..7
  b.c[1] = *(const h8*)(base + 8);  // K = kh*16+8..15
}

// Branchless packed epilogue. D value v lives at (row m0+v+8*hi, col n0+r).
// After a DPP xor-1 exchange, each lane of a column pair can build the packed
// dword: even lanes store row groups v=0..3, odd lanes v=4..7. 32 lanes x 4
// dword stores = the full 16x16 tile; no EXEC divergence, conflict-free banks.
__device__ __forceinline__ void store_h_tile(_Float16* __restrict__ hs, int m0,
                                             int hi, int r, int n0, v8f acc,
                                             float bias) {
  #pragma unroll
  for (int v = 0; v < 8; ++v) acc[v] = fmaxf(acc[v] + bias, 0.0f);
  const int odd = r & 1;
  _Float16* base = &hs[(m0 + 4 * odd + 8 * hi) * HS_ST + n0 + (r & ~1)];
  #pragma unroll
  for (int vv = 0; vv < 4; ++vv) {
    const float mineE = acc[vv];
    const float othE  = dpp_swap<0xB1>(mineE);
    const float mineO = acc[vv + 4];
    const float othO  = dpp_swap<0xB1>(mineO);
    const float lo = odd ? othO  : mineE;   // col (r&~1)
    const float hf = odd ? mineO : othE;    // col (r|1)
    h2 p;
    p.x = (_Float16)lo;
    p.y = (_Float16)hf;
    *(h2*)(base + vv * HS_ST) = p;          // row m0 + vv + 4*odd + 8*hi
  }
}

__global__ void __launch_bounds__(256)
pack_weights(const float* __restrict__ W1, const float* __restrict__ W2,
             const float* __restrict__ W3, _Float16* __restrict__ W1p,
             _Float16* __restrict__ W2p, _Float16* __restrict__ W3p) {
  const int i = blockIdx.x * 256 + threadIdx.x;
  if (i < 256 * 64) {                       // W1: (256,38) -> (256,64) zero-padded
    const int n = i >> 6, k = i & 63;
    W1p[i] = (_Float16)((k < 38) ? W1[n * 38 + k] : 0.0f);
  }
  if (i < 256 * 256) W2p[i] = (_Float16)W2[i];
  if (i < 128 * 256) W3p[i] = (_Float16)W3[i];
}

__global__ void __launch_bounds__(256)
fused_mlp(const float* __restrict__ X,
          const _Float16* __restrict__ W1p, const float* __restrict__ b1,
          const _Float16* __restrict__ W2p, const float* __restrict__ b2,
          const _Float16* __restrict__ W3p, const float* __restrict__ b3,
          Head hIncl, Head hEcc, Head hMm,
          float* __restrict__ Out) {
  extern __shared__ _Float16 lds[];
  _Float16* Xs  = lds;                       // 128 x 72 halves  (18 KB)
  _Float16* h1s = lds + TILE_M * XS_ST;      // 128 x 264 halves (66 KB)
  _Float16* h2s = h1s + TILE_M * HS_ST;      // 128 x 264 halves (66 KB)

  const int tid  = threadIdx.x;
  const int lane = tid & 31;
  const int wave = tid >> 5;
  const int r    = lane & 15;   // A: M row / B: N col / D: N col
  const int hi   = lane >> 4;   // lane half
  const int rowBase = blockIdx.x * TILE_M;

  // ---- stage X tile into LDS as f16 (K padded 38->64 with zeros), pair-packed
  for (int i = tid; i < TILE_M * 32; i += 256) {
    const int m = i >> 5, kp = (i & 31) * 2;
    const int gbase = (rowBase + m) * 38 + kp;
    h2 p;
    p.x = (_Float16)((kp     < 38) ? X[gbase]     : 0.0f);
    p.y = (_Float16)((kp + 1 < 38) ? X[gbase + 1] : 0.0f);
    *(h2*)&Xs[m * XS_ST + kp] = p;
  }
  __syncthreads();   // the ONLY wg-wide barrier: h1s/h2s strips are wave-private

  const int m0 = wave * 16;  // this wave's 16-row strip

  // ---- layer 1: h1 = relu(X @ W1^T + b1)   (M=16, N=256, K=64)
  {
    AF a[2];
    #pragma unroll
    for (int ks = 0; ks < 2; ++ks)
      load_a(a[ks], &Xs[(m0 + r) * XS_ST + ks * 32 + hi * 16]);
    #pragma unroll 2
    for (int jn = 0; jn < 16; ++jn) {
      const int n0 = jn * 16;
      v8f acc = {};
      #pragma unroll
      for (int ks = 0; ks < 2; ++ks) {
        AF b;
        load_b(b, &W1p[(n0 + r) * 64 + ks * 32 + hi * 16]);
        acc = __builtin_amdgcn_wmma_f32_16x16x32_f16(false, a[ks].v, false, b.v,
                                                     (short)0, acc, false, false);
      }
      store_h_tile(h1s, m0, hi, r, n0, acc, b1[n0 + r]);
    }
  }

  // ---- layer 2: h2 = relu(h1 @ W2^T + b2)  (M=16, N=256, K=256)
  {
    AF a[8];
    #pragma unroll
    for (int ks = 0; ks < 8; ++ks)
      load_a(a[ks], &h1s[(m0 + r) * HS_ST + ks * 32 + hi * 16]);
    #pragma unroll 2
    for (int jn = 0; jn < 16; ++jn) {
      const int n0 = jn * 16;
      v8f acc = {};
      #pragma unroll
      for (int ks = 0; ks < 8; ++ks) {
        AF b;
        load_b(b, &W2p[(n0 + r) * 256 + ks * 32 + hi * 16]);
        acc = __builtin_amdgcn_wmma_f32_16x16x32_f16(false, a[ks].v, false, b.v,
                                                     (short)0, acc, false, false);
      }
      store_h_tile(h2s, m0, hi, r, n0, acc, b2[n0 + r]);
    }
  }

  // ---- layer 3: out = h2 @ W3^T + b3  (M=16, N=128, K=256), kept in registers
  v8f outt[8];
  {
    AF a[8];
    #pragma unroll
    for (int ks = 0; ks < 8; ++ks)
      load_a(a[ks], &h2s[(m0 + r) * HS_ST + ks * 32 + hi * 16]);
    #pragma unroll 2
    for (int jn = 0; jn < 8; ++jn) {
      const int n0 = jn * 16;
      v8f acc = {};
      #pragma unroll
      for (int ks = 0; ks < 8; ++ks) {
        AF b;
        load_b(b, &W3p[(n0 + r) * 256 + ks * 32 + hi * 16]);
        acc = __builtin_amdgcn_wmma_f32_16x16x32_f16(false, a[ks].v, false, b.v,
                                                     (short)0, acc, false, false);
      }
      const float bias = b3[n0 + r];
      #pragma unroll
      for (int v = 0; v < 8; ++v) acc[v] += bias;
      outt[jn] = acc;
    }
  }

  // ---- heads: per-row dots over P=128, DPP butterfly inside each 16-lane half
  const Head heads[3] = {hIncl, hEcc, hMm};
  #pragma unroll
  for (int t = 0; t < 3; ++t) {
    const Head H = heads[t];
    float alin[8], ae[8];
    #pragma unroll
    for (int v = 0; v < 8; ++v) { alin[v] = 0.0f; ae[v] = 0.0f; }
    #pragma unroll
    for (int j = 0; j < 8; ++j) {           // N tile j, this lane owns col 16j+r
      const float lwj = H.lw[j * 16 + r];
      const float bwj = H.bew[j * 16 + r];
      #pragma unroll
      for (int v = 0; v < 8; ++v) {
        alin[v] += outt[j][v] * lwj;
        ae[v]   += outt[j][v] * bwj;
      }
    }
    float sl = 0.0f, se = 0.0f;             // cndmask selection tree, no branches
    #pragma unroll
    for (int v = 0; v < 8; ++v) {
      const float rl = reduce16(alin[v]);
      const float re = reduce16(ae[v]);
      if (r == v) { sl = rl; se = re; }
    }
    const float lb  = H.lb[0];
    const float beb = H.beb[0];
    const float bw  = H.bw[0];
    const float bb  = H.bb[0];
    if (r < 8) {                            // one lane per row writes the result
      const int row   = rowBase + m0 + r + 8 * hi;
      const float eps = X[row * 38 + 0];
      const float lin = sl + lb;
      const float e   = se * eps + beb;
      Out[row * 3 + t] = bw * e * lin + bb + X[row * 38 + H.ridx];
    }
  }
}

extern "C" void kernel_launch(void* const* d_in, const int* in_sizes, int n_in,
                              void* d_out, int out_size, void* d_ws, size_t ws_size,
                              hipStream_t stream) {
  const float* X  = (const float*)d_in[0];
  const float* W1 = (const float*)d_in[1];
  const float* b1 = (const float*)d_in[2];
  const float* W2 = (const float*)d_in[3];
  const float* b2 = (const float*)d_in[4];
  const float* W3 = (const float*)d_in[5];
  const float* b3 = (const float*)d_in[6];

  Head hs[3];
  const int ridx[3] = {7, 9, 12};  // INCL, ECC, MM residual columns
  for (int t = 0; t < 3; ++t) {
    const int base = 7 + t * 6;
    hs[t].lw   = (const float*)d_in[base + 0];
    hs[t].lb   = (const float*)d_in[base + 1];
    hs[t].bew  = (const float*)d_in[base + 2];
    hs[t].beb  = (const float*)d_in[base + 3];
    hs[t].bw   = (const float*)d_in[base + 4];
    hs[t].bb   = (const float*)d_in[base + 5];
    hs[t].ridx = ridx[t];
  }

  // f16 weight workspace: W1p 256x64, W2p 256x256, W3p 128x256 (229 KB total)
  _Float16* W1p = (_Float16*)d_ws;
  _Float16* W2p = W1p + 256 * 64;
  _Float16* W3p = W2p + 256 * 256;

  pack_weights<<<256, 256, 0, stream>>>(W1, W2, W3, W1p, W2p, W3p);

  const size_t shmem = (size_t)(TILE_M * XS_ST + 2 * TILE_M * HS_ST) * sizeof(_Float16); // 150 KB
  (void)hipFuncSetAttribute((const void*)fused_mlp,
                            hipFuncAttributeMaxDynamicSharedMemorySize, (int)shmem);

  const int nblocks = 262144 / TILE_M;  // 2048
  fused_mlp<<<nblocks, 256, shmem, stream>>>(X, W1p, b1, W2p, b2, W3p, b3,
                                             hs[0], hs[1], hs[2], (float*)d_out);
}
